// ContTimeLSTM_72602127171628
// MI455X (gfx1250) — compile-verified, matrix operand order checked by reading, not run on hardware
//
#include <hip/hip_runtime.h>
#include <hip/hip_bf16.h>
#include <stdint.h>

typedef __attribute__((ext_vector_type(16))) _Float16 v16h;
typedef __attribute__((ext_vector_type(8)))  _Float16 v8h;
typedef __attribute__((ext_vector_type(8)))  float    v8f;

#define BB    64
#define LL    512
#define DINN  512
#define HHH   1024
#define KKK   1536        // DIN + H
#define NWG   64          // one block per 16 hidden columns
#define TPB   128         // 4 waves (wave32)
#define KSTEPS 48         // KKK / 32
#define XKS   16          // DINN / 32
#define NCT   384         // 6 gates * 64 col-tiles
#define NPACK ((size_t)NCT * KSTEPS * 512)     // packed halves of W
#define WSLICE_HALF ((size_t)6 * KSTEPS * 512) // 147456 halves per block slice
#define WSLICE_BYTES (WSLICE_HALF * 2)         // 294912 B -> fits 320KB WGP LDS

// ---- workspace layout (bytes) ----
#define WPACK_OFF  ((size_t)0)
#define WPACK_SZ   (NPACK * 2)                          // 18,874,368
#define WD_OFF     ((WPACK_OFF + WPACK_SZ + 255) & ~(size_t)255)
#define WD_SZ      ((size_t)KKK * 2)
#define XF16_OFF   ((WD_OFF + WD_SZ + 255) & ~(size_t)255)
#define XF16_SZ    ((size_t)BB * LL * DINN * 2)         // 33,554,432
#define HF16_OFF   ((XF16_OFF + XF16_SZ + 255) & ~(size_t)255)
#define HF16_SZ    ((size_t)BB * HHH * 2)
#define DBUF_OFF   ((HF16_OFF + HF16_SZ + 255) & ~(size_t)255)
#define DBUF_SZ    ((size_t)BB * 4)
#define BAR_OFF    ((DBUF_OFF + DBUF_SZ + 255) & ~(size_t)255)
#define BAR_SZ     ((size_t)256)
#define WS_TOTAL   (BAR_OFF + BAR_SZ)

__device__ __forceinline__ float sigf(float x) { return 1.0f / (1.0f + __expf(-x)); }

__device__ __forceinline__ void gridBarrier(unsigned* bar) {
    __syncthreads();
    if (threadIdx.x == 0) {
        __threadfence();
        unsigned gen = __hip_atomic_load(bar + 1, __ATOMIC_RELAXED, __HIP_MEMORY_SCOPE_AGENT);
        unsigned arrived = __hip_atomic_fetch_add(bar, 1u, __ATOMIC_ACQ_REL, __HIP_MEMORY_SCOPE_AGENT);
        if (arrived == (unsigned)(NWG - 1)) {
            __hip_atomic_store(bar, 0u, __ATOMIC_RELAXED, __HIP_MEMORY_SCOPE_AGENT);
            __hip_atomic_fetch_add(bar + 1, 1u, __ATOMIC_RELEASE, __HIP_MEMORY_SCOPE_AGENT);
        } else {
            while (__hip_atomic_load(bar + 1, __ATOMIC_ACQUIRE, __HIP_MEMORY_SCOPE_AGENT) == gen) {
                __builtin_amdgcn_s_sleep(1);
            }
        }
    }
    __syncthreads();
}

// Pack W (fp32, [K=1536, 1024] per gate) into WMMA-B fragment-native f16 layout.
// Slice order: cg-major (contiguous 288KB per block), then ks-major with the 6
// gate fragments of one k-step contiguous (6KB) so DS loads use one base VGPR +
// immediate offsets. fragment = 32 lanes x 16 halves (1KB): lane L covers
// N = L%16, halves e cover K = ks*32 + (L>=16?8:0) + e + (e>=8?8:0).
__global__ void ctlstm_repack_w(const float* __restrict__ Wi, const float* __restrict__ Wf,
                                const float* __restrict__ Wie, const float* __restrict__ Wfe,
                                const float* __restrict__ Wz, const float* __restrict__ Wo,
                                _Float16* __restrict__ wpack) {
    size_t idx = (size_t)blockIdx.x * blockDim.x + threadIdx.x;
    if (idx >= NPACK) return;
    int e512 = (int)(idx & 511);
    size_t q = idx >> 9;
    int g    = (int)(q % 6);
    int ks   = (int)((q / 6) % KSTEPS);
    int cg   = (int)(q / ((size_t)6 * KSTEPS));
    int laneL = e512 >> 4;
    int e     = e512 & 15;
    int N  = laneL & 15;
    int K  = ks * 32 + ((laneL >= 16) ? 8 : 0) + e + ((e >= 8) ? 8 : 0);
    int col = cg * 16 + N;
    const float* W = (g == 0) ? Wi : (g == 1) ? Wf : (g == 2) ? Wie
                   : (g == 3) ? Wfe : (g == 4) ? Wz : Wo;
    wpack[idx] = (_Float16)W[(size_t)K * HHH + col];
}

__global__ void ctlstm_convert_x(const float* __restrict__ x, _Float16* __restrict__ xf16) {
    size_t idx = (size_t)blockIdx.x * blockDim.x + threadIdx.x;
    if (idx < (size_t)BB * LL * DINN) xf16[idx] = (_Float16)x[idx];
}

__global__ void ctlstm_init_misc(const float* __restrict__ Wd, _Float16* __restrict__ wd,
                                 float* __restrict__ dbuf, unsigned* __restrict__ bar) {
    int idx = blockIdx.x * blockDim.x + threadIdx.x;
    if (idx < KKK) wd[idx] = (_Float16)Wd[idx];
    if (idx < BB)  dbuf[idx] = 0.0f;
    if (idx < 64)  bar[idx] = 0u;
}

// GEMM micro-step: one k-step (K=32) x 6 gates. Forces all 6 B fragments into
// distinct registers so the 12 ds_load_b128 batch up and overlap the WMMA chain.
__device__ __forceinline__ void gemm_kstep(const _Float16* __restrict__ ap,
                                           const _Float16* __restrict__ lp,
                                           v8f acc[6]) {
    v8h alo = *(const v8h*)(ap);
    v8h ahi = *(const v8h*)(ap + 16);
    v16h a;
#pragma unroll
    for (int i = 0; i < 8; ++i) { a[i] = alo[i]; a[i + 8] = ahi[i]; }
    v16h bf[6];
#pragma unroll
    for (int g = 0; g < 6; ++g) bf[g] = *(const v16h*)(lp + g * 512);
#pragma unroll
    for (int g = 0; g < 6; ++g)
        acc[g] = __builtin_amdgcn_wmma_f32_16x16x32_f16(
            false, a, false, bf[g], (short)0, acc[g], false, false);
}

// Persistent recurrent kernel: 64 blocks x 128 threads (4 wave32).
// Block cg owns hidden columns [cg*16, cg*16+16); its 288KB weight slice lives in
// LDS for the whole 512-step recurrence (async-preloaded once). Wave w owns batch
// rows [16w, 16w+16) and keeps cs/ce/o for its 16x16 tile in registers (WMMA C
// layout). 288 v_wmma_f32_16x16x32_f16 per wave per step, B operands from LDS.
__global__ void __launch_bounds__(TPB)
ctlstm_persistent(const float* __restrict__ ts,
                  const _Float16* __restrict__ wpack, const _Float16* __restrict__ wd,
                  const _Float16* __restrict__ xf16, _Float16* __restrict__ hf16,
                  float* __restrict__ dbuf, unsigned* __restrict__ bar,
                  const float* __restrict__ b_i, const float* __restrict__ b_f,
                  const float* __restrict__ b_ie, const float* __restrict__ b_fe,
                  const float* __restrict__ b_z, const float* __restrict__ b_o,
                  const float* __restrict__ b_d, const float* __restrict__ beta,
                  float* __restrict__ out) {
    __shared__ float sred[TPB];
    extern __shared__ _Float16 smemw[];   // 294912 B dynamic LDS: weight slice

    const int lane  = threadIdx.x & 31;
    const int w     = threadIdx.x >> 5;       // wave id -> row tile
    const int cg    = blockIdx.x;             // column group
    const int nlo   = lane & 15;
    const int lhi   = (lane >> 4) & 1;        // lanes 16..31 -> M += 8 / K += 8
    const int col   = cg * 16 + nlo;          // hidden column owned by this lane
    const int bbase = w * 16 + lhi * 8;       // C-layout batch row base (reg r -> bbase+r)
    const int arow  = w * 16 + nlo;           // A-layout row for this lane
    const int k8    = lhi * 8;                // A-layout K sub-offset

    // ---- async preload of this block's contiguous 288KB weight slice into LDS ----
    {
        const char* gsrc = (const char*)(wpack + (size_t)cg * WSLICE_HALF);
        // Low 32 bits of a generic LDS pointer are the LDS byte offset.
        unsigned lbase = (unsigned)(uintptr_t)(void*)smemw;
        const int CHUNKS = (int)(WSLICE_BYTES / ((size_t)TPB * 16));   // 144
        for (int c = 0; c < CHUNKS; ++c) {
            unsigned off = (unsigned)(c * TPB + threadIdx.x) * 16u;
            unsigned laddr = lbase + off;
            unsigned long long gaddr = (unsigned long long)(uintptr_t)(gsrc + off);
            asm volatile("global_load_async_to_lds_b128 %0, %1, off"
                         :: "v"(laddr), "v"(gaddr) : "memory");
        }
        asm volatile("s_wait_asynccnt 0x0" ::: "memory");
        __syncthreads();
    }

    float biasv[6];
    {
        const float* bp[6] = {b_i, b_f, b_ie, b_fe, b_z, b_o};
#pragma unroll
        for (int g = 0; g < 6; ++g) biasv[g] = bp[g][col];
    }

    v8f cs = {}, ce = {}, og = {};

    for (int t = 0; t < LL; ++t) {
        // ---- phase A: h_t from carry, publish h (f16) + output (f32) ----
        v8f cvec;
#pragma unroll
        for (int r = 0; r < 8; ++r) {
            int b = bbase + r;
            float tv = ts[b * LL + t];
            float tp = (t > 0) ? ts[b * LL + t - 1] : 0.0f;
            float dp = dbuf[b];
            float c = cs[r] + (ce[r] - cs[r]) * __expf(dp * (tp - tv));
            float h = og[r] * tanhf(c);
            cvec[r] = c;
            out[((size_t)b * LL + t) * HHH + col] = h;
            hf16[(size_t)b * HHH + col] = (_Float16)h;
        }
        gridBarrier(bar);

        // ---- phase B: pre = [x_t, h_t] @ W_all + b via WMMA; B from LDS ----
        v8f acc[6];
#pragma unroll
        for (int g = 0; g < 6; ++g)
#pragma unroll
            for (int r = 0; r < 8; ++r) acc[g][r] = biasv[g];

        {
            const _Float16* lp = smemw + lane * 16;          // ks-major, 6KB per ks
            const _Float16* ap = xf16 + ((size_t)arow * LL + t) * DINN + k8;
            for (int ks = 0; ks < XKS; ++ks) {               // x part of K
                gemm_kstep(ap, lp, acc);
                ap += 32;
                lp += 6 * 512;
            }
            const _Float16* ah = hf16 + (size_t)arow * HHH + k8;
            for (int ks = XKS; ks < KSTEPS; ++ks) {          // h part of K
                gemm_kstep(ah, lp, acc);
                ah += 32;
                lp += 6 * 512;
            }
        }

        // ---- gates + state update (in C-layout registers) ----
#pragma unroll
        for (int r = 0; r < 8; ++r) {
            float ig  = sigf(acc[0][r]);
            float fg  = sigf(acc[1][r]);
            float ieg = sigf(acc[2][r]);
            float feg = sigf(acc[3][r]);
            float z   = 2.0f * sigf(acc[4][r]) - 1.0f;
            float ogr = sigf(acc[5][r]);
            cs[r] = fg * cvec[r] + ig * z;
            ce[r] = feg * ce[r] + ieg * z;
            og[r] = ogr;
        }

        // ---- block 0: decay column d (K=1536 dot per batch) ----
        if (blockIdx.x == 0) {
            int tid = threadIdx.x;
            int b = tid & 63, half = tid >> 6;
            float p = 0.0f;
            int kbeg = half * (KKK / 2), kend = kbeg + (KKK / 2);
            for (int k = kbeg; k < kend; ++k) {
                float av = (k < DINN) ? (float)xf16[((size_t)b * LL + t) * DINN + k]
                                      : (float)hf16[(size_t)b * HHH + (k - DINN)];
                p += av * (float)wd[k];
            }
            sred[tid] = p;
            __syncthreads();
            if (tid < BB) {
                float pre = sred[tid] + sred[tid + 64] + b_d[0];
                float bt = beta[0];
                float bx = bt * pre;
                float sp = (bx > 20.0f) ? bx : log1pf(__expf(bx));
                dbuf[tid] = sp / bt;
            }
            __syncthreads();
        }
        gridBarrier(bar);
    }
}

extern "C" void kernel_launch(void* const* d_in, const int* in_sizes, int n_in,
                              void* d_out, int out_size, void* d_ws, size_t ws_size,
                              hipStream_t stream) {
    (void)in_sizes; (void)n_in; (void)out_size;
    if (ws_size < WS_TOTAL) return;

    const float* x    = (const float*)d_in[0];
    const float* ts   = (const float*)d_in[1];
    const float* W_i  = (const float*)d_in[2];
    const float* b_i  = (const float*)d_in[3];
    const float* W_f  = (const float*)d_in[4];
    const float* b_f  = (const float*)d_in[5];
    const float* W_ie = (const float*)d_in[6];
    const float* b_ie = (const float*)d_in[7];
    const float* W_fe = (const float*)d_in[8];
    const float* b_fe = (const float*)d_in[9];
    const float* W_z  = (const float*)d_in[10];
    const float* b_z  = (const float*)d_in[11];
    const float* W_o  = (const float*)d_in[12];
    const float* b_o  = (const float*)d_in[13];
    const float* W_d  = (const float*)d_in[14];
    const float* b_d  = (const float*)d_in[15];
    const float* beta = (const float*)d_in[16];

    char* ws = (char*)d_ws;
    _Float16* wpack = (_Float16*)(ws + WPACK_OFF);
    _Float16* wd    = (_Float16*)(ws + WD_OFF);
    _Float16* xf16  = (_Float16*)(ws + XF16_OFF);
    _Float16* hf16  = (_Float16*)(ws + HF16_OFF);
    float*    dbuf  = (float*)(ws + DBUF_OFF);
    unsigned* bar   = (unsigned*)(ws + BAR_OFF);

    ctlstm_repack_w<<<(unsigned)((NPACK + 255) / 256), 256, 0, stream>>>(
        W_i, W_f, W_ie, W_fe, W_z, W_o, wpack);
    ctlstm_convert_x<<<(unsigned)(((size_t)BB * LL * DINN + 255) / 256), 256, 0, stream>>>(
        x, xf16);
    ctlstm_init_misc<<<8, 256, 0, stream>>>(W_d, wd, dbuf, bar);
    ctlstm_persistent<<<NWG, TPB, (unsigned)WSLICE_BYTES, stream>>>(
        ts, wpack, wd, xf16, hf16, dbuf, bar,
        b_i, b_f, b_ie, b_fe, b_z, b_o, b_d, beta, (float*)d_out);
}
